// Attention_4312147165333
// MI455X (gfx1250) — compile-verified, matrix operand order checked by reading, not run on hardware
//
#include <hip/hip_runtime.h>
#include <hip/hip_bf16.h>
#include <math.h>

typedef __attribute__((ext_vector_type(16))) __bf16 v16bf;
typedef __attribute__((ext_vector_type(8)))  float  v8f;
typedef unsigned int uint32x4 __attribute__((ext_vector_type(4)));
typedef int          int32x4  __attribute__((ext_vector_type(4)));
typedef int          int32x8  __attribute__((ext_vector_type(8)));

#define BB 2
#define LL 2048
#define DD 1024
#define HH 16
#define DH 64
#define NEGBIG (-1.0e9f)
#define LROW 68   /* LDS row stride in floats: 64 data + 4 pad (bank spread) */

static __device__ __forceinline__ v8f wmma_bf16(v16bf a, v16bf b, v8f c) {
  // D = A(16x32 bf16) * B(32x16 bf16) + C(16x16 f32)
  return __builtin_amdgcn_wmma_f32_16x16x32_bf16(false, a, false, b, (short)0, c,
                                                 false, false);
}

static __device__ __forceinline__ v16bf pack16(float4 f0, float4 f1, float4 f2, float4 f3) {
  v16bf r;
  r[0]=(__bf16)f0.x;  r[1]=(__bf16)f0.y;  r[2]=(__bf16)f0.z;  r[3]=(__bf16)f0.w;
  r[4]=(__bf16)f1.x;  r[5]=(__bf16)f1.y;  r[6]=(__bf16)f1.z;  r[7]=(__bf16)f1.w;
  r[8]=(__bf16)f2.x;  r[9]=(__bf16)f2.y;  r[10]=(__bf16)f2.z; r[11]=(__bf16)f2.w;
  r[12]=(__bf16)f3.x; r[13]=(__bf16)f3.y; r[14]=(__bf16)f3.z; r[15]=(__bf16)f3.w;
  return r;
}

// Lane's 16 bf16 elements of a 16-bit WMMA A/B operand, K-run of 32 contiguous
// at p.  kLo = (lane<16)?0:8.  e<8 -> K=kLo+e ; e>=8 -> K=16+kLo+(e-8).
static __device__ __forceinline__ v16bf load_row32(const float* p, int kLo) {
  float4 f0 = *(const float4*)(p + kLo);
  float4 f1 = *(const float4*)(p + kLo + 4);
  float4 f2 = *(const float4*)(p + 16 + kLo);
  float4 f3 = *(const float4*)(p + 16 + kLo + 4);
  return pack16(f0, f1, f2, f3);
}

// ---------------------------------------------------------------------------
// TDM: async DMA of a 2-D f32 tile (tile_w x tile_h) global -> LDS, with
// +4-DWORD padding after every 64 DWORDs (LDS row stride = 68 floats).
// Rows past rows_left are zero-filled by the TDM OOB logic.
// D# fields per CDNA5 ISA ch.8; groups 2/3 unused for 2-D tensors.
// clang-23 toolchain: 6-arg builtin (g0, g1, g2, g3, extra, cpol).
// ---------------------------------------------------------------------------
static __device__ __forceinline__ void tdm_load_tile_f32(
    const float* gsrc, unsigned lds_byte, unsigned rows_left,
    unsigned row_stride_elems, unsigned tile_w, unsigned tile_h) {
  unsigned long long ga = (unsigned long long)(uintptr_t)gsrc;
  uint32x4 g0 = { 1u,                                   // count=1 (valid D#)
                  lds_byte,                             // lds_addr
                  (unsigned)(ga & 0xFFFFFFFFu),
                  (unsigned)((ga >> 32) & 0x01FFFFFFull) | (2u << 30) }; // type=2
  int32x8 g1 = {
      (int)((2u << 16) |        // data_size = 4 bytes
            (1u << 20) |        // pad_enable
            (5u << 22) |        // pad_interval: 64 DWORDs
            (3u << 25)),        // pad_amount:   4 DWORDs
      (int)((tile_w & 0xFFFFu) << 16),                        // tensor_dim0 lo
      (int)((rows_left & 0xFFFFu) << 16),                     // tensor_dim1 lo
      (int)(((rows_left >> 16) & 0xFFFFu) | ((tile_w & 0xFFFFu) << 16)), // dim1 hi | tile_dim0
      (int)(tile_h & 0xFFFFu),                                // tile_dim1 (tile_dim2=0)
      (int)row_stride_elems,                                  // tensor_dim0_stride lo
      0, 0 };
  int32x4 z4 = { 0, 0, 0, 0 };
  int32x8 z8 = { 0, 0, 0, 0, 0, 0, 0, 0 };
  __builtin_amdgcn_tensor_load_to_lds(g0, g1, z4, z4, z8, 0);
}

static __device__ __forceinline__ unsigned lds_off(const void* p) {
  return (unsigned)(uintptr_t)p;   // low 32 bits of flat addr = LDS byte offset
}

// ---------------------------------------------------------------------------
// Kernel 1: flash attention.  Block = 4 waves = 64 q-rows of one (b,h); all
// waves share TDM-staged K/V tiles (32 keys x 64 dh, f32) with double buffer.
// ---------------------------------------------------------------------------
__global__ __launch_bounds__(128) void attn_fa_kernel(
    const float* __restrict__ Q, const float* __restrict__ K,
    const float* __restrict__ V, float* __restrict__ AO) {
  __shared__ float kv[2][2][32 * LROW];   // [buf][K/V][tile]  (34.8 KB)
  __shared__ float plds[4][16 * 32];      // per-wave P transpose (8 KB)

  const int lane = threadIdx.x & 31;
  const int wave = threadIdx.x >> 5;
  const int qb = blockIdx.x & 31;                 // 32 q-blocks of 64 rows
  const int h  = (blockIdx.x >> 5) & 15;
  const int b  = blockIdx.x >> 9;
  const int qBase = qb << 6;
  const int q0 = qBase + wave * 16;               // this wave's 16 q rows

  const int n16  = lane & 15;
  const int half = lane >> 4;
  const int kLo  = half << 3;

  // Q A-operands for this wave (rows q0..q0+15, dh in two K=32 chunks)
  const float* qrow = Q + ((size_t)(b * LL + q0 + n16)) * DD + h * DH;
  v16bf qa0 = load_row32(qrow, kLo);
  v16bf qa1 = load_row32(qrow + 32, kLo);

  v8f acc0 = {}, acc1 = {}, acc2 = {}, acc3 = {};
  float mrow[8], lrow[8];
#pragma unroll
  for (int r = 0; r < 8; ++r) { mrow[r] = -INFINITY; lrow[r] = 0.0f; }

  float* lp = &plds[wave][0];
  const int nblocks = (qBase >> 5) + 2;           // key blocks of 32, uniform

  if (wave == 0) {                                // preload block 0
    const float* kg = K + ((size_t)(b * LL)) * DD + h * DH;
    const float* vg = V + ((size_t)(b * LL)) * DD + h * DH;
    tdm_load_tile_f32(kg, lds_off(&kv[0][0][0]), LL, DD, DH, 32);
    tdm_load_tile_f32(vg, lds_off(&kv[0][1][0]), LL, DD, DH, 32);
  }

  for (int i = 0; i < nblocks; ++i) {
    const int cur = i & 1;
    const int kb  = i << 5;
    if (wave == 0) {
      if (i + 1 < nblocks) {
        const int kn = kb + 32;
        const float* kg = K + ((size_t)(b * LL + kn)) * DD + h * DH;
        const float* vg = V + ((size_t)(b * LL + kn)) * DD + h * DH;
        tdm_load_tile_f32(kg, lds_off(&kv[1 - cur][0][0]), LL - kn, DD, DH, 32);
        tdm_load_tile_f32(vg, lds_off(&kv[1 - cur][1][0]), LL - kn, DD, DH, 32);
        __builtin_amdgcn_s_wait_tensorcnt(2);     // current pair complete
      } else {
        __builtin_amdgcn_s_wait_tensorcnt(0);
      }
    }
    __syncthreads();                              // kv[cur] ready for all waves
    const float* kbuf = &kv[cur][0][0];
    const float* vbuf = &kv[cur][1][0];

    // ---- S = Q * K^T (two 16-key subtiles, K-dim = dh = 64) ----
    v8f s[2];
#pragma unroll
    for (int sub = 0; sub < 2; ++sub) {
      const float* kr = kbuf + (sub * 16 + n16) * LROW;
      v16bf kb0 = load_row32(kr, kLo);
      v16bf kb1 = load_row32(kr + 32, kLo);
      v8f c = {};
      c = wmma_bf16(qa0, kb0, c);
      c = wmma_bf16(qa1, kb1, c);
      s[sub] = c;
    }
    // ---- scale + causal mask (matches w*m + (-1e9)*(1-m)) ----
#pragma unroll
    for (int sub = 0; sub < 2; ++sub) {
      int key = kb + sub * 16 + n16;
#pragma unroll
      for (int r = 0; r < 8; ++r) {
        int qi = q0 + r + half * 8;
        float v = s[sub][r] * 0.125f;             // 1/sqrt(64)
        s[sub][r] = (key > qi) ? NEGBIG : v;
      }
    }
    // ---- online softmax (row reductions across 16-lane halves) ----
#pragma unroll
    for (int r = 0; r < 8; ++r) {
      float mx = fmaxf(s[0][r], s[1][r]);
      mx = fmaxf(mx, __shfl_xor(mx, 1, 32));
      mx = fmaxf(mx, __shfl_xor(mx, 2, 32));
      mx = fmaxf(mx, __shfl_xor(mx, 4, 32));
      mx = fmaxf(mx, __shfl_xor(mx, 8, 32));
      float mnew  = fmaxf(mrow[r], mx);
      float alpha = __expf(mrow[r] - mnew);
      float p0 = __expf(s[0][r] - mnew);
      float p1 = __expf(s[1][r] - mnew);
      s[0][r] = p0; s[1][r] = p1;
      float rs = p0 + p1;
      rs += __shfl_xor(rs, 1, 32);
      rs += __shfl_xor(rs, 2, 32);
      rs += __shfl_xor(rs, 4, 32);
      rs += __shfl_xor(rs, 8, 32);
      lrow[r] = lrow[r] * alpha + rs;
      mrow[r] = mnew;
      acc0[r] *= alpha; acc1[r] *= alpha; acc2[r] *= alpha; acc3[r] *= alpha;
    }
    // ---- transpose P from C-layout to A-layout via per-wave LDS ----
#pragma unroll
    for (int sub = 0; sub < 2; ++sub)
#pragma unroll
      for (int r = 0; r < 8; ++r)
        lp[(r + half * 8) * 32 + sub * 16 + n16] = s[sub][r];
    asm volatile("s_wait_dscnt 0" ::: "memory");
    v16bf pa;
#pragma unroll
    for (int e = 0; e < 16; ++e) {
      int k = (e < 8) ? (kLo + e) : (16 + kLo + (e - 8));
      pa[e] = (__bf16)lp[n16 * 32 + k];
    }
    asm volatile("s_wait_dscnt 0" ::: "memory");
    // ---- O += P(16x32) * V(32x64) from LDS (conflict-free via pad) ----
#pragma unroll
    for (int nt = 0; nt < 4; ++nt) {
      int colLocal = nt * 16 + n16;
      v16bf vb;
#pragma unroll
      for (int e = 0; e < 16; ++e) {
        int k = (e < 8) ? (kLo + e) : (16 + kLo + (e - 8));
        vb[e] = (__bf16)vbuf[k * LROW + colLocal];
      }
      if      (nt == 0) acc0 = wmma_bf16(pa, vb, acc0);
      else if (nt == 1) acc1 = wmma_bf16(pa, vb, acc1);
      else if (nt == 2) acc2 = wmma_bf16(pa, vb, acc2);
      else              acc3 = wmma_bf16(pa, vb, acc3);
    }
    __syncthreads();            // all reads of kv[cur] done before overwrite
  }
  // ---- normalize and store merged-head output [B, L, D] ----
#pragma unroll
  for (int r = 0; r < 8; ++r) {
    float inv = 1.0f / lrow[r];
    int row = q0 + r + half * 8;
    size_t base = ((size_t)(b * LL + row)) * DD + h * DH + n16;
    AO[base + 0]  = acc0[r] * inv;
    AO[base + 16] = acc1[r] * inv;
    AO[base + 32] = acc2[r] * inv;
    AO[base + 48] = acc3[r] * inv;
  }
}

// ---------------------------------------------------------------------------
// Kernel 2: out = A[4096,1024] @ W[1024,1024] + bias.
// Block = 8 waves sharing one 64-col j-tile; W tile (32x64 f32) TDM-staged
// into LDS per K-step, double-buffered.  Wave = 16x64 output tile.
// ---------------------------------------------------------------------------
__global__ __launch_bounds__(256) void proj_kernel(
    const float* __restrict__ A, const float* __restrict__ W,
    const float* __restrict__ bias, float* __restrict__ out) {
  __shared__ float wlds[2][32 * LROW];   // 17.4 KB

  const int lane = threadIdx.x & 31;
  const int wave = threadIdx.x >> 5;
  const int j0 = (blockIdx.x & 15) << 6;           // 16 col tiles of 64
  const int m0 = ((blockIdx.x >> 4) << 7) + wave * 16;  // 32 row groups of 128
  const int n16  = lane & 15;
  const int half = lane >> 4;
  const int kLo  = half << 3;

  v8f acc0 = {}, acc1 = {}, acc2 = {}, acc3 = {};
  const float* arow = A + (size_t)(m0 + n16) * DD;

  if (wave == 0)
    tdm_load_tile_f32(W + j0, lds_off(&wlds[0][0]), DD, DD, DH, 32);

  const int nsteps = DD / 32;
  for (int i = 0; i < nsteps; ++i) {
    const int cur = i & 1;
    const int kd  = i << 5;
    if (wave == 0) {
      if (i + 1 < nsteps) {
        tdm_load_tile_f32(W + (size_t)(kd + 32) * DD + j0,
                          lds_off(&wlds[1 - cur][0]), DD - (kd + 32), DD, DH, 32);
        __builtin_amdgcn_s_wait_tensorcnt(1);
      } else {
        __builtin_amdgcn_s_wait_tensorcnt(0);
      }
    }
    __syncthreads();
    const float* wl = &wlds[cur][0];

    v16bf aa = load_row32(arow + kd, kLo);
    v16bf w0, w1, w2, w3;
#pragma unroll
    for (int e = 0; e < 16; ++e) {
      int k = (e < 8) ? (kLo + e) : (16 + kLo + (e - 8));
      const float* wr = wl + k * LROW + n16;
      w0[e] = (__bf16)wr[0];
      w1[e] = (__bf16)wr[16];
      w2[e] = (__bf16)wr[32];
      w3[e] = (__bf16)wr[48];
    }
    acc0 = wmma_bf16(aa, w0, acc0);
    acc1 = wmma_bf16(aa, w1, acc1);
    acc2 = wmma_bf16(aa, w2, acc2);
    acc3 = wmma_bf16(aa, w3, acc3);
    __syncthreads();
  }
  float b0 = bias[j0 + n16],      b1 = bias[j0 + 16 + n16];
  float b2 = bias[j0 + 32 + n16], b3 = bias[j0 + 48 + n16];
#pragma unroll
  for (int r = 0; r < 8; ++r) {
    int row = m0 + r + half * 8;
    size_t base = (size_t)row * DD + j0 + n16;
    out[base + 0]  = acc0[r] + b0;
    out[base + 16] = acc1[r] + b1;
    out[base + 32] = acc2[r] + b2;
    out[base + 48] = acc3[r] + b3;
  }
}

extern "C" void kernel_launch(void* const* d_in, const int* in_sizes, int n_in,
                              void* d_out, int out_size, void* d_ws, size_t ws_size,
                              hipStream_t stream) {
  (void)in_sizes; (void)n_in; (void)out_size; (void)ws_size;
  const float* Q    = (const float*)d_in[0];
  const float* K    = (const float*)d_in[1];
  const float* V    = (const float*)d_in[2];
  const float* W    = (const float*)d_in[3];
  const float* bias = (const float*)d_in[4];
  // d_in[5] = n_head scalar (16), compile-time constant here.
  float* AO = (float*)d_ws;                    // [B, L, D] fp32 = 16.8 MB

  // K1: (b,h) x 32 q-blocks of 64 rows, 4 waves/block sharing K/V tiles.
  attn_fa_kernel<<<1024, 128, 0, stream>>>(Q, K, V, AO);
  // K2: 16 j-tiles x 32 m-groups, 8 waves/block sharing the W tile.
  proj_kernel<<<512, 256, 0, stream>>>(AO, W, bias, (float*)d_out);
}
